// MultiHeadSelfAttention_79224966742751
// MI455X (gfx1250) — matrix operand order repaired and checked by reading, not compiled, not run on hardware
//
#include <hip/hip_runtime.h>
#include <hip/hip_bf16.h>

// MI455X (gfx1250) CDNA5: bf16 WMMA flash attention.
// Block = 8 waves sharing one (b,h); K/V tiles double-buffered in LDS via the
// Tensor Data Mover (with D# row padding for bank-conflict-free reads).
// ws layout (bf16): xb[4M] | wqb[1M] | wkb[1M] | wvb[1M] | wob[1M] |
//                   Q[4M] | K[4M] | Vt[4M] | O[4M]   => 48 MiB total.

#define D_MODEL   1024
#define NUM_HEADS 16
#define HEAD_DIM  64
#define SEQ       2048
#define BATCH     2

#define KT_STRIDE 66   // 64 bf16 + 1-dword TDM pad
#define VT_STRIDE 34   // 32 bf16 + 1-dword TDM pad

typedef __attribute__((ext_vector_type(16))) __bf16 v16bf;
typedef __attribute__((ext_vector_type(8)))  float  v8f;
typedef __attribute__((ext_vector_type(4)))  unsigned int u32x4;
typedef __attribute__((ext_vector_type(8)))  int i32x8;
typedef __attribute__((ext_vector_type(4)))  int i32x4;

#if defined(__has_builtin)
#if __has_builtin(__builtin_amdgcn_tensor_load_to_lds) && \
    __has_builtin(__builtin_amdgcn_s_wait_tensorcnt)
#define HAVE_TDM 1
#endif
#endif

__device__ __forceinline__ v8f wmma_bf(v16bf a, v16bf b, v8f c) {
  return __builtin_amdgcn_wmma_f32_16x16x32_bf16(false, a, false, b, (short)0, c,
                                                 false, false);
}

// ---- WMMA fragment loaders (ISA 7.12.2, wave32) --------------------------
// A (16x32): lane holds row m=lane&15; element e -> k=e+16*(lane>>4) (contiguous)
__device__ __forceinline__ v16bf load_a_bf16(const __bf16* __restrict__ base,
                                             int ld, int row0, int k0) {
  const int lane = threadIdx.x & 31;
  const __bf16* p = base + (size_t)(row0 + (lane & 15)) * ld + k0 + (lane >> 4) * 16;
  v16bf a;
#pragma unroll
  for (int e = 0; e < 8; ++e) a[e] = p[e];
#pragma unroll
  for (int e = 0; e < 8; ++e) a[e + 8] = p[e + 8];
  return a;
}
// B (32x16) from N-row-major source (B[k][n] = src[n0+n][k]):
// lane holds col n=lane&15; element e -> k = e + 16*(lane>>4) (contiguous)
__device__ __forceinline__ v16bf load_b_bf16(const __bf16* __restrict__ base,
                                             int ld, int n0, int k0) {
  const int lane = threadIdx.x & 31;
  const __bf16* p = base + (size_t)(n0 + (lane & 15)) * ld + k0 + (lane >> 4) * 16;
  v16bf b;
#pragma unroll
  for (int e = 0; e < 16; ++e) b[e] = p[e];
  return b;
}

#ifdef HAVE_TDM
// 2D bf16 tile load via Tensor Data Mover (D# per ISA 08_async_tensor §8.3/8.4).
// pad_code: if pad_en, insert 1 dword of LDS padding every 2^(pad_code+1) dwords.
__device__ __forceinline__ void tdm_load_2d_bf16(unsigned lds_off, const void* gptr,
                                                 unsigned tile_x, unsigned tile_y,
                                                 unsigned tensor_x, unsigned tensor_y,
                                                 unsigned stride_x,
                                                 unsigned pad_en, unsigned pad_code) {
  const unsigned long long ga = (unsigned long long)(size_t)gptr;
  u32x4 g0;
  g0[0] = 1u;  // count=1 valid descriptor, user mode
  g0[1] = (unsigned)__builtin_amdgcn_readfirstlane((int)lds_off);
  g0[2] = (unsigned)__builtin_amdgcn_readfirstlane((int)(ga & 0xFFFFFFFFu));
  g0[3] = (unsigned)__builtin_amdgcn_readfirstlane(
              (int)(((ga >> 32) & 0x1FFFFFFu) | (2u << 30)));  // type=2 (image)
  i32x8 g1;
  g1[0] = (int)((1u << 16) | (pad_en << 20) | (pad_code << 22));  // data_size=2B
  g1[1] = (int)((tensor_x & 0xFFFFu) << 16);                      // dim0 lo16
  g1[2] = (int)(((tensor_x >> 16) & 0xFFFFu) | ((tensor_y & 0xFFFFu) << 16));
  g1[3] = (int)(((tensor_y >> 16) & 0xFFFFu) | ((tile_x & 0xFFFFu) << 16));
  g1[4] = (int)(tile_y & 0xFFFFu);  // tile_dim1; tile_dim2=0
  g1[5] = (int)stride_x;            // tensor_dim0_stride (lo32)
  g1[6] = 0;
  g1[7] = 0;
  i32x4 z4 = {0, 0, 0, 0};
  i32x8 z8 = {0, 0, 0, 0, 0, 0, 0, 0};
  __builtin_amdgcn_tensor_load_to_lds(g0, g1, z4, z4, z8, 0);
}
#endif

// ---- Kernel 0: f32 -> bf16 conversion ------------------------------------
__global__ __launch_bounds__(256) void cvt_bf16_kernel(const float* __restrict__ src,
                                                       __bf16* __restrict__ dst,
                                                       int n) {
  const int i = blockIdx.x * blockDim.x + threadIdx.x;
  if (i < n) dst[i] = (__bf16)src[i];
}

// ---- Kernel 1: QKV projection (out = x @ W^T), all-bf16 operands ---------
__global__ __launch_bounds__(256) void qkv_gemm_kernel(
    const __bf16* __restrict__ xb,
    const __bf16* __restrict__ wqb, const __bf16* __restrict__ wkb,
    const __bf16* __restrict__ wvb,
    __bf16* __restrict__ Q, __bf16* __restrict__ K, __bf16* __restrict__ Vt) {
  const int widx = threadIdx.x >> 5;
  const int lane = threadIdx.x & 31;
  const int tile = blockIdx.x * 8 + widx;   // 0..4095
  const int mt = tile & 255;
  const int ng = tile >> 8;
  const int which = blockIdx.y;
  const __bf16* W = (which == 0) ? wqb : (which == 1) ? wkb : wvb;
  const int row0 = mt * 16;
  const int col0 = ng * 64;

  v8f acc[4] = {};
  // 2x-unrolled double buffer: no register rotation copies.
  v16bf a0 = load_a_bf16(xb, D_MODEL, row0, 0);
  v16bf b0[4], b1[4];
#pragma unroll
  for (int j = 0; j < 4; ++j) b0[j] = load_b_bf16(W, D_MODEL, col0 + j * 16, 0);

  for (int kk = 0; kk < D_MODEL; kk += 64) {
    __builtin_prefetch(xb + (size_t)row0 * D_MODEL + kk + 64, 0, 1);
    v16bf a1 = load_a_bf16(xb, D_MODEL, row0, kk + 32);
#pragma unroll
    for (int j = 0; j < 4; ++j) b1[j] = load_b_bf16(W, D_MODEL, col0 + j * 16, kk + 32);
#pragma unroll
    for (int j = 0; j < 4; ++j) acc[j] = wmma_bf(a0, b0[j], acc[j]);
    if (kk + 64 < D_MODEL) {
      a0 = load_a_bf16(xb, D_MODEL, row0, kk + 64);
#pragma unroll
      for (int j = 0; j < 4; ++j) b0[j] = load_b_bf16(W, D_MODEL, col0 + j * 16, kk + 64);
    }
#pragma unroll
    for (int j = 0; j < 4; ++j) acc[j] = wmma_bf(a1, b1[j], acc[j]);
  }

  const int n = lane & 15, hf = lane >> 4;
#pragma unroll
  for (int j = 0; j < 4; ++j) {
#pragma unroll
    for (int g = 0; g < 8; ++g) {
      const int r = row0 + g + hf * 8;       // b*SEQ + s
      const int c = col0 + j * 16 + n;       // h*64 + d
      const int b = r >> 11, s = r & (SEQ - 1);
      const int h = c >> 6, d = c & 63;
      const __bf16 val = (__bf16)acc[j][g];
      if (which == 0)
        Q[((size_t)(b * NUM_HEADS + h) * SEQ + s) * HEAD_DIM + d] = val;
      else if (which == 1)
        K[((size_t)(b * NUM_HEADS + h) * SEQ + s) * HEAD_DIM + d] = val;
      else
        Vt[((size_t)(b * NUM_HEADS + h) * HEAD_DIM + d) * SEQ + s] = val;
    }
  }
}

// ---- Kernel 2: RoPE on Q,K (pairs (2i,2i+1) of head dim) -----------------
__global__ __launch_bounds__(256) void rope_kernel(__bf16* __restrict__ Q,
                                                   __bf16* __restrict__ K,
                                                   const int* __restrict__ pos) {
  const int idx = blockIdx.x * blockDim.x + threadIdx.x;
  const int i = idx & 31;
  const int bhs = idx >> 5;
  const int s = bhs & (SEQ - 1);
  const float p = (float)pos[s];
  const float ang = p * __expf(-(float)i * (9.2103403719761836f / 32.0f));
  const float cs = __cosf(ang), sn = __sinf(ang);
  const size_t base = (size_t)bhs * HEAD_DIM + 2 * i;
  float e = (float)Q[base], o = (float)Q[base + 1];
  Q[base]     = (__bf16)(e * cs - o * sn);
  Q[base + 1] = (__bf16)(e * sn + o * cs);
  e = (float)K[base]; o = (float)K[base + 1];
  K[base]     = (__bf16)(e * cs - o * sn);
  K[base + 1] = (__bf16)(e * sn + o * cs);
}

// ---- Kernel 3: causal flash attention ------------------------------------
// Block = 8 waves on 8 consecutive q-tiles of one (b,h). Wave 0 streams the
// shared 32-key K tile and 64x32 V tile through the TDM, double-buffered.
__global__ __launch_bounds__(256) void attn_kernel(const __bf16* __restrict__ Q,
                                                   const __bf16* __restrict__ K,
                                                   const __bf16* __restrict__ Vt,
                                                   __bf16* __restrict__ O) {
  __shared__ __bf16 KtL[2][32 * KT_STRIDE];  // keys x head_dim (padded rows)
  __shared__ __bf16 VtL[2][64 * VT_STRIDE];  // head_dim x keys (padded rows)
  __shared__ __bf16 Pbuf[8][16][32];         // per-wave P staging

  const int widx = threadIdx.x >> 5;
  const int lane = threadIdx.x & 31;
  const int qtb = blockIdx.x & 15;           // q-tile group (8 tiles each)
  const int bh  = blockIdx.x >> 4;           // b*16 + h
  const int b = bh >> 4, h = bh & 15;
  const int q0 = (qtb * 8 + widx) * 16;
  const int myLim = q0 + 16;                 // keys this wave needs
  const int blockLim = (qtb + 1) * 128;      // keys the block needs
  const __bf16* Qbh = Q + (size_t)bh * SEQ * HEAD_DIM;
  const __bf16* Kbh = K + (size_t)bh * SEQ * HEAD_DIM;
  const __bf16* Vbh = Vt + (size_t)bh * HEAD_DIM * SEQ;
  const int n = lane & 15, hf = lane >> 4;

  const v16bf aQ0 = load_a_bf16(Qbh, HEAD_DIM, q0, 0);
  const v16bf aQ1 = load_a_bf16(Qbh, HEAD_DIM, q0, 32);

  v8f accO[4] = {};
  float mrow[8], lrow[8];
#pragma unroll
  for (int g = 0; g < 8; ++g) { mrow[g] = -__builtin_inff(); lrow[g] = 0.f; }

#ifdef HAVE_TDM
  const unsigned ktL[2] = {(unsigned)(size_t)&KtL[0][0], (unsigned)(size_t)&KtL[1][0]};
  const unsigned vtL[2] = {(unsigned)(size_t)&VtL[0][0], (unsigned)(size_t)&VtL[1][0]};
  if (widx == 0) {  // prologue: stage tiles for kb=0 into buffer 0
    tdm_load_2d_bf16(ktL[0], Kbh, HEAD_DIM, 32, HEAD_DIM, SEQ, HEAD_DIM, 1, 4);
    tdm_load_2d_bf16(vtL[0], Vbh, 32, HEAD_DIM, SEQ, HEAD_DIM, SEQ, 1, 3);
  }
#endif

  for (int kb = 0; kb < blockLim; kb += 32) {
    const int buf = (kb >> 5) & 1;
    __syncthreads();  // all readers of buf^1 finished (previous iteration)
#ifdef HAVE_TDM
    if (widx == 0) {
      if (kb + 32 < blockLim) {
        tdm_load_2d_bf16(ktL[buf ^ 1], Kbh + (size_t)(kb + 32) * HEAD_DIM,
                         HEAD_DIM, 32, HEAD_DIM, SEQ, HEAD_DIM, 1, 4);
        tdm_load_2d_bf16(vtL[buf ^ 1], Vbh + (kb + 32),
                         32, HEAD_DIM, SEQ, HEAD_DIM, SEQ, 1, 3);
        __builtin_amdgcn_s_wait_tensorcnt(2);  // current pair complete
      } else {
        __builtin_amdgcn_s_wait_tensorcnt(0);
      }
    }
#endif
    __syncthreads();  // buffer `buf` ready for all waves

    if (kb < myLim) {  // wave-uniform causal guard (EXEC stays all-ones)
      v16bf bK[4];
#ifdef HAVE_TDM
#pragma unroll
      for (int t = 0; t < 4; ++t) {  // t = {ksub 0/16} x {k0 0/32}
        const int ksub = (t >> 1) * 16, k0 = (t & 1) * 32;
        const __bf16* p = &KtL[buf][(ksub + n) * KT_STRIDE + k0 + hf * 16];
#pragma unroll
        for (int e = 0; e < 16; ++e) bK[t][e] = p[e];
      }
#else
      bK[0] = load_b_bf16(Kbh, HEAD_DIM, kb, 0);
      bK[1] = load_b_bf16(Kbh, HEAD_DIM, kb, 32);
      bK[2] = load_b_bf16(Kbh, HEAD_DIM, kb + 16, 0);
      bK[3] = load_b_bf16(Kbh, HEAD_DIM, kb + 16, 32);
#endif
      v8f s0 = {}, s1 = {};
      s0 = wmma_bf(aQ0, bK[0], s0);
      s0 = wmma_bf(aQ1, bK[1], s0);
      s1 = wmma_bf(aQ0, bK[2], s1);
      s1 = wmma_bf(aQ1, bK[3], s1);
#pragma unroll
      for (int g = 0; g < 8; ++g) {
        const int qi = q0 + g + hf * 8;
        float v0 = (kb + n      <= qi) ? s0[g] * 0.125f : -__builtin_inff();
        float v1 = (kb + 16 + n <= qi) ? s1[g] * 0.125f : -__builtin_inff();
        float mx = fmaxf(v0, v1);
#pragma unroll
        for (int msk = 1; msk < 16; msk <<= 1)
          mx = fmaxf(mx, __shfl_xor(mx, msk, 32));
        const float mnew = fmaxf(mrow[g], mx);
        const float alpha = __expf(mrow[g] - mnew);
        const float p0 = __expf(v0 - mnew);
        const float p1 = __expf(v1 - mnew);
        float ps = p0 + p1;
#pragma unroll
        for (int msk = 1; msk < 16; msk <<= 1) ps += __shfl_xor(ps, msk, 32);
        lrow[g] = lrow[g] * alpha + ps;
        mrow[g] = mnew;
#pragma unroll
        for (int nt = 0; nt < 4; ++nt) accO[nt][g] *= alpha;
        const int mr = g + hf * 8;
        Pbuf[widx][mr][n] = (__bf16)p0;
        Pbuf[widx][mr][16 + n] = (__bf16)p1;
      }
      asm volatile("s_wait_dscnt 0" ::: "memory");  // wave-private P staging
      v16bf aP;
      {
        const int m = lane & 15;
#pragma unroll
        for (int e = 0; e < 16; ++e) {
          const int k = e + hf * 16;
          aP[e] = Pbuf[widx][m][k];
        }
      }
#pragma unroll
      for (int nt = 0; nt < 4; ++nt) {
        v16bf bV;
#ifdef HAVE_TDM
        const __bf16* p = &VtL[buf][(nt * 16 + n) * VT_STRIDE + hf * 16];
#else
        const __bf16* p = Vbh + (size_t)(nt * 16 + n) * SEQ + kb + hf * 16;
#endif
#pragma unroll
        for (int e = 0; e < 16; ++e) bV[e] = p[e];
        accO[nt] = wmma_bf(aP, bV, accO[nt]);
      }
    }
  }

#pragma unroll
  for (int nt = 0; nt < 4; ++nt) {
#pragma unroll
    for (int g = 0; g < 8; ++g) {
      const int srow = q0 + g + hf * 8;
      const float val = accO[nt][g] / lrow[g];
      O[((size_t)b * SEQ + srow) * D_MODEL + h * HEAD_DIM + nt * 16 + n] =
          (__bf16)val;
    }
  }
}

// ---- Kernel 4: output projection (out = O @ Wo^T), bf16 -> f32 -----------
__global__ __launch_bounds__(256) void out_gemm_kernel(const __bf16* __restrict__ O,
                                                       const __bf16* __restrict__ Wob,
                                                       float* __restrict__ out) {
  const int widx = threadIdx.x >> 5;
  const int lane = threadIdx.x & 31;
  const int tile = blockIdx.x * 8 + widx;
  const int mt = tile & 255;
  const int ng = tile >> 8;
  const int row0 = mt * 16, col0 = ng * 64;

  v8f acc[4] = {};
  v16bf a0 = load_a_bf16(O, D_MODEL, row0, 0);
  v16bf b0[4], b1[4];
#pragma unroll
  for (int j = 0; j < 4; ++j) b0[j] = load_b_bf16(Wob, D_MODEL, col0 + j * 16, 0);

  for (int kk = 0; kk < D_MODEL; kk += 64) {
    v16bf a1 = load_a_bf16(O, D_MODEL, row0, kk + 32);
#pragma unroll
    for (int j = 0; j < 4; ++j) b1[j] = load_b_bf16(Wob, D_MODEL, col0 + j * 16, kk + 32);
#pragma unroll
    for (int j = 0; j < 4; ++j) acc[j] = wmma_bf(a0, b0[j], acc[j]);
    if (kk + 64 < D_MODEL) {
      a0 = load_a_bf16(O, D_MODEL, row0, kk + 64);
#pragma unroll
      for (int j = 0; j < 4; ++j) b0[j] = load_b_bf16(Wob, D_MODEL, col0 + j * 16, kk + 64);
    }
#pragma unroll
    for (int j = 0; j < 4; ++j) acc[j] = wmma_bf(a1, b1[j], acc[j]);
  }

  const int n = lane & 15, hf = lane >> 4;
#pragma unroll
  for (int j = 0; j < 4; ++j)
#pragma unroll
    for (int g = 0; g < 8; ++g)
      out[(size_t)(row0 + g + hf * 8) * D_MODEL + col0 + j * 16 + n] = acc[j][g];
}

extern "C" void kernel_launch(void* const* d_in, const int* in_sizes, int n_in,
                              void* d_out, int out_size, void* d_ws, size_t ws_size,
                              hipStream_t stream) {
  const float* x  = (const float*)d_in[0];
  const float* wq = (const float*)d_in[1];
  const float* wk = (const float*)d_in[2];
  const float* wv = (const float*)d_in[3];
  const float* wo = (const float*)d_in[4];
  const int* pos  = (const int*)d_in[5];
  float* out = (float*)d_out;

  const size_t NELEM = (size_t)BATCH * SEQ * D_MODEL;  // 4M
  const size_t WELEM = (size_t)D_MODEL * D_MODEL;      // 1M
  __bf16* xb  = (__bf16*)d_ws;
  __bf16* wqb = xb + NELEM;
  __bf16* wkb = wqb + WELEM;
  __bf16* wvb = wkb + WELEM;
  __bf16* wob = wvb + WELEM;
  __bf16* Q   = wob + WELEM;
  __bf16* K   = Q + NELEM;
  __bf16* Vt  = K + NELEM;
  __bf16* O   = Vt + NELEM;

  cvt_bf16_kernel<<<(int)(NELEM / 256), 256, 0, stream>>>(x, xb, (int)NELEM);
  cvt_bf16_kernel<<<(int)(WELEM / 256), 256, 0, stream>>>(wq, wqb, (int)WELEM);
  cvt_bf16_kernel<<<(int)(WELEM / 256), 256, 0, stream>>>(wk, wkb, (int)WELEM);
  cvt_bf16_kernel<<<(int)(WELEM / 256), 256, 0, stream>>>(wv, wvb, (int)WELEM);
  cvt_bf16_kernel<<<(int)(WELEM / 256), 256, 0, stream>>>(wo, wob, (int)WELEM);

  qkv_gemm_kernel<<<dim3(512, 3), 256, 0, stream>>>(xb, wqb, wkb, wvb, Q, K, Vt);
  rope_kernel<<<(BATCH * NUM_HEADS * SEQ * 32) / 256, 256, 0, stream>>>(Q, K, pos);
  attn_kernel<<<512, 256, 0, stream>>>(Q, K, Vt, O);
  out_gemm_kernel<<<512, 256, 0, stream>>>(O, wob, out);
}
